// TokenMudule_53506702573964
// MI455X (gfx1250) — matrix operand order repaired and checked by reading, not compiled, not run on hardware
//
#include <hip/hip_runtime.h>
#include <cmath>

// ---------------- problem constants ----------------
#define BDIM 32
#define HWDIM 196
#define HSIDE 14
#define CDIM 768
#define NCDIM 4096
#define MROWS (BDIM * HWDIM)   // 6272

typedef __attribute__((ext_vector_type(16))) __bf16 v16bf;
typedef __attribute__((ext_vector_type(8)))  __bf16 v8bf;
typedef __attribute__((ext_vector_type(8)))  float  v8f;

// ---------------------------------------------------------------------------
// Kernel 1: SpaceSelfAware + per-(b,c,ki,kj) L2 norm over hw + GeM(3x3) pool
//           + residual.  One workgroup (256 thr, 8 waves) per (b,c) pair.
//           Column of 196 values staged into LDS via async global->LDS DMA.
// ---------------------------------------------------------------------------
__global__ void ssa_kernel(const float* __restrict__ x, float* __restrict__ t) {
    const int b = blockIdx.x / CDIM;
    const int c = blockIdx.x % CDIM;
    __shared__ float col[HWDIM];
    __shared__ float part[9][8];
    __shared__ float norms[9];

    const int tid  = threadIdx.x;
    const int lane = tid & 31;
    const int w    = tid >> 5;
    const bool active = tid < HWDIM;

    // ---- async load of the strided column x[b, :, c] into LDS ----
    {
        unsigned long long sb = (unsigned long long)(x + (size_t)b * HWDIM * CDIM + c);
        if (active) {
            unsigned ldsAddr = (unsigned)(uintptr_t)(&col[tid]);
            unsigned voff    = (unsigned)(tid * CDIM * 4);
            asm volatile("global_load_async_to_lds_b32 %0, %1, %2"
                         :: "v"(ldsAddr), "v"(voff), "s"(sb) : "memory");
        }
    }
    asm volatile("s_wait_asynccnt 0x0" ::: "memory");
    __syncthreads();

    const int i = tid / HSIDE;
    const int j = tid % HSIDE;
    const float center = active ? col[tid] : 0.0f;

    float p[9];
#pragma unroll
    for (int kk = 0; kk < 9; ++kk) {
        const int ki = kk / 3, kj = kk % 3;
        const int ri = i + ki - 2;          // top pad = 2
        const int rj = j + kj - 1;          // left pad = 1
        float v = 0.0f;
        if (active && ri >= 0 && ri < HSIDE && rj >= 0 && rj < HSIDE)
            v = col[ri * HSIDE + rj];
        p[kk] = center * v;
        float s = p[kk] * p[kk];
#pragma unroll
        for (int off = 16; off; off >>= 1) s += __shfl_xor(s, off, 32);
        if (lane == 0) part[kk][w] = s;
    }
    __syncthreads();
    if (tid < 9) {
        float s = 0.0f;
#pragma unroll
        for (int ww = 0; ww < 8; ++ww) s += part[tid][ww];
        norms[tid] = fmaxf(sqrtf(s), 1e-12f);
    }
    __syncthreads();

    if (active) {
        float acc = 0.0f;
#pragma unroll
        for (int kk = 0; kk < 9; ++kk) {
            float v = fmaxf(p[kk] / norms[kk], 1e-6f);
            acc += v * v * v;
        }
        const float pooled = cbrtf(acc * (1.0f / 9.0f));
        t[((size_t)b * HWDIM + tid) * CDIM + c] = pooled + center;
    }
}

// ---------------------------------------------------------------------------
// Kernel 2: per-token L2 normalize over channels, emit bf16 activations.
// ---------------------------------------------------------------------------
__global__ void rownorm_kernel(const float* __restrict__ t, __bf16* __restrict__ xb) {
    const int r   = blockIdx.x;
    const int tid = threadIdx.x;
    const int lane = tid & 31, w = tid >> 5;
    const float* row = t + (size_t)r * CDIM;
    float v0 = row[tid], v1 = row[tid + 256], v2 = row[tid + 512];
    float s = v0 * v0 + v1 * v1 + v2 * v2;
#pragma unroll
    for (int off = 16; off; off >>= 1) s += __shfl_xor(s, off, 32);
    __shared__ float part[8];
    __shared__ float nrm;
    if (lane == 0) part[w] = s;
    __syncthreads();
    if (tid == 0) {
        float tot = 0.0f;
#pragma unroll
        for (int ww = 0; ww < 8; ++ww) tot += part[ww];
        nrm = fmaxf(sqrtf(tot), 1e-12f);
    }
    __syncthreads();
    const float n = nrm;
    __bf16* orow = xb + (size_t)r * CDIM;
    orow[tid]       = (__bf16)(v0 / n);
    orow[tid + 256] = (__bf16)(v1 / n);
    orow[tid + 512] = (__bf16)(v2 / n);
}

// ---------------------------------------------------------------------------
// Kernel 3: transpose-convert fp32 K x N weights -> bf16 N x K (col-major B)
//           so WMMA B fragments are contiguous per lane.
// ---------------------------------------------------------------------------
__global__ void wconv_kernel(const float* __restrict__ wsrc, __bf16* __restrict__ wt,
                             int K, int N) {
    __shared__ float tile[32][33];
    const int n0 = blockIdx.x * 32;
    const int k0 = blockIdx.y * 32;
    const int tx = threadIdx.x, ty = threadIdx.y;   // 32 x 8
#pragma unroll
    for (int r = 0; r < 32; r += 8)
        tile[ty + r][tx] = wsrc[(size_t)(k0 + ty + r) * N + n0 + tx];
    __syncthreads();
#pragma unroll
    for (int r = 0; r < 32; r += 8)
        wt[(size_t)(n0 + ty + r) * K + k0 + tx] = (__bf16)tile[tx][ty + r];
}

// ---------------------------------------------------------------------------
// WMMA fragment loaders (bf16 16x16x32 layouts, wave32).
// A 16x32: lanes 0-15 row m, K k..k+7 and k+16..k+23; lanes 16-31 get +8.
// B 32x16 (from N x K transposed weights): lanes 0-15 K k..k+7 and k+16..k+23; 16-31 +8.
// ---------------------------------------------------------------------------
__device__ inline v16bf load_frag_a(const __bf16* __restrict__ base, int ldk,
                                    int row, int k, int lane) {
    const int sub = (lane >> 4) & 1;
    const __bf16* p = base + (size_t)row * ldk + k + sub * 8;
    v8bf lo = *(const v8bf*)(p);
    v8bf hi = *(const v8bf*)(p + 16);
    return __builtin_shufflevector(lo, hi, 0, 1, 2, 3, 4, 5, 6, 7,
                                           8, 9, 10, 11, 12, 13, 14, 15);
}
__device__ inline v16bf load_frag_b(const __bf16* __restrict__ base, int ldk,
                                    int col, int k, int lane) {
    const int sub = (lane >> 4) & 1;
    const __bf16* p = base + (size_t)col * ldk + k + sub * 8;
    v8bf lo = *(const v8bf*)(p);
    v8bf hi = *(const v8bf*)(p + 16);
    return __builtin_shufflevector(lo, hi, 0, 1, 2, 3, 4, 5, 6, 7,
                                           8, 9, 10, 11, 12, 13, 14, 15);
}

// ---------------------------------------------------------------------------
// Kernel 4: WMMA GEMM  C[M,N] = A[M,K](bf16) * Bt[N,K](bf16)^T + bias,
//           optional exact-erf GELU, fp32 or bf16 output.
//           8 waves per WG, each wave computes a 32x32 tile (2x2 WMMA tiles).
// ---------------------------------------------------------------------------
template <bool GELU, bool OUTBF16>
__global__ void gemm_kernel(const __bf16* __restrict__ A, const __bf16* __restrict__ Bt,
                            const float* __restrict__ bias, void* __restrict__ Cout,
                            int M, int N, int K) {
    const int tid  = threadIdx.x;
    const int lane = tid & 31;
    const int w    = tid >> 5;
    const int wm   = w & 1;
    const int wn   = w >> 1;
    const int m0   = blockIdx.y * 64 + wm * 32;
    const int n0   = blockIdx.x * 128 + wn * 32;
    const int l15  = lane & 15;

    v8f acc00 = {}, acc01 = {}, acc10 = {}, acc11 = {};

    for (int k = 0; k < K; k += 32) {
        if (k + 64 < K)
            __builtin_prefetch(A + (size_t)min(m0 + l15, M - 1) * K + k + 64, 0, 1);
        const v16bf a0 = load_frag_a(A, K, min(m0 + l15,      M - 1), k, lane);
        const v16bf a1 = load_frag_a(A, K, min(m0 + 16 + l15, M - 1), k, lane);
        const v16bf b0 = load_frag_b(Bt, K, n0 + l15,      k, lane);
        const v16bf b1 = load_frag_b(Bt, K, n0 + 16 + l15, k, lane);
        acc00 = __builtin_amdgcn_wmma_f32_16x16x32_bf16(false, a0, false, b0,
                                                        (short)0, acc00, false, false);
        acc01 = __builtin_amdgcn_wmma_f32_16x16x32_bf16(false, a0, false, b1,
                                                        (short)0, acc01, false, false);
        acc10 = __builtin_amdgcn_wmma_f32_16x16x32_bf16(false, a1, false, b0,
                                                        (short)0, acc10, false, false);
        acc11 = __builtin_amdgcn_wmma_f32_16x16x32_bf16(false, a1, false, b1,
                                                        (short)0, acc11, false, false);
    }

    // C/D layout: VGPR r -> row (r + 8*(lane>=16)), col = lane&15
    const int rbase = (lane >> 4) << 3;
    const v8f* accs[2][2] = {{&acc00, &acc01}, {&acc10, &acc11}};
#pragma unroll
    for (int mt = 0; mt < 2; ++mt) {
#pragma unroll
        for (int nt = 0; nt < 2; ++nt) {
            const int col = n0 + nt * 16 + l15;
            const float bv = bias[col];
            const v8f a = *accs[mt][nt];
#pragma unroll
            for (int r = 0; r < 8; ++r) {
                const int row = m0 + mt * 16 + rbase + r;
                if (row < M) {
                    float v = a[r] + bv;
                    if (GELU) v = 0.5f * v * (1.0f + erff(v * 0.70710678118654752f));
                    if (OUTBF16)
                        ((__bf16*)Cout)[(size_t)row * N + col] = (__bf16)v;
                    else
                        ((float*)Cout)[(size_t)row * N + col] = v;
                }
            }
        }
    }
}

// ---------------------------------------------------------------------------
// Kernel 5: GeM(p=3) over hw per (b,c), emit bf16 g[32,768].
// ---------------------------------------------------------------------------
__global__ void gempool_kernel(const float* __restrict__ t2, __bf16* __restrict__ gb) {
    const int b   = blockIdx.x;
    const int tid = threadIdx.x;
    const float* base = t2 + (size_t)b * HWDIM * CDIM;
    float a0 = 0.0f, a1 = 0.0f, a2 = 0.0f;
    for (int hw = 0; hw < HWDIM; ++hw) {
        const float* row = base + (size_t)hw * CDIM;
        float v0 = fmaxf(row[tid],       1e-6f); a0 += v0 * v0 * v0;
        float v1 = fmaxf(row[tid + 256], 1e-6f); a1 += v1 * v1 * v1;
        float v2 = fmaxf(row[tid + 512], 1e-6f); a2 += v2 * v2 * v2;
    }
    __bf16* g = gb + (size_t)b * CDIM;
    g[tid]       = (__bf16)cbrtf(a0 * (1.0f / HWDIM));
    g[tid + 256] = (__bf16)cbrtf(a1 * (1.0f / HWDIM));
    g[tid + 512] = (__bf16)cbrtf(a2 * (1.0f / HWDIM));
}

// ---------------------------------------------------------------------------
// Kernel 6: final per-row L2 normalize of the 4096-d descriptor.
// ---------------------------------------------------------------------------
__global__ void headnorm_kernel(const float* __restrict__ desc, float* __restrict__ out) {
    const int b   = blockIdx.x;
    const int tid = threadIdx.x;
    const int lane = tid & 31, w = tid >> 5;
    const float* row = desc + (size_t)b * NCDIM;
    float s = 0.0f;
    for (int i = tid; i < NCDIM; i += 256) { float v = row[i]; s += v * v; }
#pragma unroll
    for (int off = 16; off; off >>= 1) s += __shfl_xor(s, off, 32);
    __shared__ float part[8];
    __shared__ float nrm;
    if (lane == 0) part[w] = s;
    __syncthreads();
    if (tid == 0) {
        float tot = 0.0f;
#pragma unroll
        for (int ww = 0; ww < 8; ++ww) tot += part[ww];
        nrm = fmaxf(sqrtf(tot), 1e-12f);
    }
    __syncthreads();
    const float n = nrm;
    for (int i = tid; i < NCDIM; i += 256) out[(size_t)b * NCDIM + i] = row[i] / n;
}

// ---------------------------------------------------------------------------
// Host-side launch
// ---------------------------------------------------------------------------
extern "C" void kernel_launch(void* const* d_in, const int* in_sizes, int n_in,
                              void* d_out, int out_size, void* d_ws, size_t ws_size,
                              hipStream_t stream) {
    const float* x      = (const float*)d_in[0];
    const float* fc1_w  = (const float*)d_in[1];
    const float* fc1_b  = (const float*)d_in[2];
    const float* fc2_w  = (const float*)d_in[3];
    const float* fc2_b  = (const float*)d_in[4];
    const float* head_w = (const float*)d_in[5];
    const float* head_b = (const float*)d_in[6];
    float* out = (float*)d_out;

    char* ws = (char*)d_ws;
    constexpr size_t sz_t  = (size_t)MROWS * CDIM * 4;
    constexpr size_t sz_xb = (size_t)MROWS * CDIM * 2;
    constexpr size_t off_t  = 0;
    constexpr size_t off_t2 = off_t + sz_t;
    constexpr size_t off_X  = off_t2 + sz_t;
    constexpr size_t off_H  = off_X + sz_xb;
    constexpr size_t off_W1 = off_H + sz_xb;
    constexpr size_t off_W2 = off_W1 + (size_t)CDIM * CDIM * 2;
    constexpr size_t off_Wh = off_W2 + (size_t)CDIM * CDIM * 2;
    constexpr size_t off_g  = off_Wh + (size_t)CDIM * NCDIM * 2;
    constexpr size_t off_d  = off_g + (size_t)BDIM * CDIM * 2;

    float*  t    = (float*)(ws + off_t);
    float*  t2   = (float*)(ws + off_t2);
    __bf16* Xbf  = (__bf16*)(ws + off_X);
    __bf16* Hbf  = (__bf16*)(ws + off_H);
    __bf16* W1t  = (__bf16*)(ws + off_W1);
    __bf16* W2t  = (__bf16*)(ws + off_W2);
    __bf16* Wht  = (__bf16*)(ws + off_Wh);
    __bf16* gbf  = (__bf16*)(ws + off_g);
    float*  desc = (float*)(ws + off_d);

    // 1) SSA + space fusion + residual
    ssa_kernel<<<BDIM * CDIM, 256, 0, stream>>>(x, t);
    // 2) channel L2 norm -> bf16 activations
    rownorm_kernel<<<MROWS, 256, 0, stream>>>(t, Xbf);
    // 3) weight transpose+convert (bf16, N x K)
    dim3 tb(32, 8);
    wconv_kernel<<<dim3(CDIM / 32, CDIM / 32), tb, 0, stream>>>(fc1_w, W1t, CDIM, CDIM);
    wconv_kernel<<<dim3(CDIM / 32, CDIM / 32), tb, 0, stream>>>(fc2_w, W2t, CDIM, CDIM);
    wconv_kernel<<<dim3(NCDIM / 32, CDIM / 32), tb, 0, stream>>>(head_w, Wht, CDIM, NCDIM);
    // 4) MLP: fc1 + GELU (bf16 out), fc2 (fp32 out)
    gemm_kernel<true,  true ><<<dim3(CDIM / 128, MROWS / 64), 256, 0, stream>>>(
        Xbf, W1t, fc1_b, Hbf, MROWS, CDIM, CDIM);
    gemm_kernel<false, false><<<dim3(CDIM / 128, MROWS / 64), 256, 0, stream>>>(
        Hbf, W2t, fc2_b, t2, MROWS, CDIM, CDIM);
    // 5) GeM head pool
    gempool_kernel<<<BDIM, 256, 0, stream>>>(t2, gbf);
    // 6) head GEMM (M=32 -> one row-block, clamped/guarded)
    gemm_kernel<false, false><<<dim3(NCDIM / 128, 1), 256, 0, stream>>>(
        gbf, Wht, head_b, desc, BDIM, NCDIM, CDIM);
    // 7) final L2 normalize
    headnorm_kernel<<<BDIM, 256, 0, stream>>>(desc, out);
}
